// ParticleNet_75565654606024
// MI455X (gfx1250) — matrix operand order repaired and checked by reading, not compiled, not run on hardware
//
#include <hip/hip_runtime.h>
#include <hip/hip_bf16.h>

// ParticleNet for MI455X (gfx1250).
// Compute-bound (~0.56 TFLOP of channel GEMMs, activations fit in 192MB L2),
// so the edge convs run on v_wmma_f32_16x16x32_f16 with BN folded into f16
// weights (f32 accumulate). One wave32 per point: the K=16 neighbor rows map
// exactly onto WMMA M=16.

#define BB 512
#define PP 125
#define FF 16
#define KK 16
#define BN_EPS 1e-3f

typedef _Float16 v16h __attribute__((ext_vector_type(16)));
typedef _Float16 v8h  __attribute__((ext_vector_type(8)));
typedef float    v8f  __attribute__((ext_vector_type(8)));

// ---------------------------------------------------------------------------
// Weight prep: fold BN into W (per-output-channel scale) and bias, emit f16.
// frag==1: store in WMMA B-fragment order so each lane's 16 halves are one
// contiguous 32B chunk:  offset = ((nt*(Kd/32)+kc)*32 + lane)*16 + j
// where lane = (k%32<16 ? n%16 : 16 + n%16), j = k%16.
// ---------------------------------------------------------------------------
__global__ void prep_weights(const float* __restrict__ W,
                             const float* __restrict__ beta,
                             const float* __restrict__ gamma,
                             const float* __restrict__ mean,
                             const float* __restrict__ var,
                             _Float16* __restrict__ Wout,
                             float* __restrict__ bout,
                             int Kd, int N, int frag) {
  const int total = Kd * N;
  for (int t = blockIdx.x * blockDim.x + threadIdx.x; t < total;
       t += gridDim.x * blockDim.x) {
    const int k = t / N, n = t - k * N;
    const float s = gamma[n] * rsqrtf(var[n] + BN_EPS);
    const float v = W[t] * s;
    size_t o;
    if (frag) {
      const int nt = n >> 4, nl = n & 15, kc = k >> 5, kin = k & 31;
      const int lane = (kin < 16) ? nl : (nl + 16);
      const int j = kin & 15;
      o = ((size_t)(nt * (Kd >> 5) + kc) * 32 + lane) * 16 + j;
    } else {
      o = (size_t)t;
    }
    Wout[o] = (_Float16)v;
    if (t < N) {
      const float st = gamma[t] * rsqrtf(var[t] + BN_EPS);
      bout[t] = beta[t] - mean[t] * st;
    }
  }
}

// ---------------------------------------------------------------------------
// Input stage: points from (r,theta), mask = any(f != 0), fts0 = BN(features)
// ---------------------------------------------------------------------------
__global__ void input_kernel(const float* __restrict__ feat,
                             const float* __restrict__ beta,
                             const float* __restrict__ gamma,
                             const float* __restrict__ mean,
                             const float* __restrict__ var,
                             float* __restrict__ pts,
                             float* __restrict__ mask,
                             float* __restrict__ fts0) {
  const int pid = blockIdx.x * blockDim.x + threadIdx.x;
  if (pid >= BB * PP) return;
  const float* f = feat + (size_t)pid * FF;
  float fl[FF];
  float any = 0.f;
#pragma unroll
  for (int c = 0; c < FF; ++c) {
    fl[c] = f[c];
    if (fl[c] != 0.f) any = 1.f;
  }
  mask[pid] = any;
  pts[pid * 2 + 0] = fl[0] * cosf(fl[1]);
  pts[pid * 2 + 1] = fl[0] * sinf(fl[1]);
#pragma unroll
  for (int c = 0; c < FF; ++c)
    fts0[(size_t)pid * FF + c] =
        (fl[c] - mean[c]) * rsqrtf(var[c] + BN_EPS) * gamma[c] + beta[c];
}

// ---------------------------------------------------------------------------
// kNN: one workgroup (8 waves) per batch. Coords (+1e9 shift on masked pts)
// staged in LDS. Per wave: 125 distances (4 q per lane), then 17 iterations of
// tie-stable (d, then index) wave min-reduction == top_k(-d, 17); drop first.
// ---------------------------------------------------------------------------
template <int CC>
__global__ void __launch_bounds__(256) knn_kernel(const float* __restrict__ src,
                                                  const float* __restrict__ mask,
                                                  int* __restrict__ idxo) {
  extern __shared__ char smem[];
  float* cds = (float*)smem;       // PP*CC shifted coords
  float* rsq = cds + PP * CC;      // PP squared norms
  const int b = blockIdx.x;
  const int tid = threadIdx.x;
  for (int t = tid; t < PP * CC; t += 256) {
    const int p = t / CC, c = t - p * CC;
    cds[t] = src[((size_t)b * PP + p) * CC + c] + 1e9f * (1.0f - mask[b * PP + p]);
  }
  __syncthreads();
  for (int p = tid; p < PP; p += 256) {
    float s = 0.f;
    for (int c = 0; c < CC; ++c) {
      const float v = cds[p * CC + c];
      s += v * v;
    }
    rsq[p] = s;
  }
  __syncthreads();
  const int wave = tid >> 5, lane = tid & 31;
  for (int p = wave; p < PP; p += 8) {
    float d[4];
    int qv[4];
#pragma unroll
    for (int j = 0; j < 4; ++j) {
      const int q = lane + j * 32;
      if (q < PP) {
        float dot = 0.f;
        for (int c = 0; c < CC; ++c) dot += cds[p * CC + c] * cds[q * CC + c];
        d[j] = rsq[p] - 2.f * dot + rsq[q];
        qv[j] = q;
      } else {
        d[j] = 3.4e38f;
        qv[j] = 1 << 30;
      }
    }
    for (int t = 0; t < KK + 1; ++t) {
      float bd = 3.4e38f;
      int bq = 1 << 30;
#pragma unroll
      for (int j = 0; j < 4; ++j)
        if (d[j] < bd || (d[j] == bd && qv[j] < bq)) { bd = d[j]; bq = qv[j]; }
      for (int off = 16; off > 0; off >>= 1) {
        const float od = __shfl_xor(bd, off, 32);
        const int   oq = __shfl_xor(bq, off, 32);
        if (od < bd || (od == bd && oq < bq)) { bd = od; bq = oq; }
      }
      if (t > 0 && lane == 0) idxo[((size_t)b * PP + p) * KK + (t - 1)] = bq;
#pragma unroll
      for (int j = 0; j < 4; ++j)
        if (qv[j] == bq) { d[j] = 3.4e38f; qv[j] = 1 << 30; }
    }
  }
}

// ---------------------------------------------------------------------------
// WMMA helpers. A (16x32 f16, contiguous K): lanes 0-15 hold row=lane,
// K {0..15}; lanes 16-31 hold row=lane-16, K {16..31}.
// ---------------------------------------------------------------------------
template <int Kd>
__device__ __forceinline__ v16h load_a_frag(const _Float16* Xs, int mrow,
                                            int hi, int kc) {
  const size_t ro = (size_t)mrow * Kd + (size_t)kc * 32;
  const v8h lo = *(const v8h*)(Xs + ro + hi * 16);
  const v8h hh = *(const v8h*)(Xs + ro + hi * 16 + 8);
  v16h a;
#pragma unroll
  for (int j = 0; j < 8; ++j) {
    a[j] = lo[j];
    a[8 + j] = hh[j];
  }
  return a;
}

// One conv layer: H = relu(X @ W + bias), X is 16xKd f16 in LDS, W is
// fragment-ordered f16 in global, H written 16xC f16 to LDS.
template <int Kd, int C>
__device__ __forceinline__ void conv_wmma(const _Float16* Xs,
                                          const _Float16* __restrict__ Wg,
                                          const float* __restrict__ bias,
                                          _Float16* Hs, int lane) {
  constexpr int KC = Kd >> 5;
  const int mrow = lane & 15;
  const int hi = lane >> 4;
  for (int nt = 0; nt < C / 16; ++nt) {
    v8f acc = {0.f, 0.f, 0.f, 0.f, 0.f, 0.f, 0.f, 0.f};
    for (int kc = 0; kc < KC; ++kc) {
      const v16h a = load_a_frag<Kd>(Xs, mrow, hi, kc);
      const v16h bm = *(const v16h*)(Wg + ((size_t)(nt * KC + kc) * 32 + lane) * 16);
      acc = __builtin_amdgcn_wmma_f32_16x16x32_f16(false, a, false, bm,
                                                   (short)0, acc, false, false);
    }
    const int n = nt * 16 + mrow;
    const float bn = bias[n];
#pragma unroll
    for (int i = 0; i < 8; ++i) {
      float v = acc[i] + bn;
      v = v > 0.f ? v : 0.f;
      // C/D layout: vgpr i -> m = i + 8*hi, n = nt*16 + (lane&15)
      Hs[(size_t)(i + hi * 8) * C + n] = (_Float16)v;
    }
  }
}

// ---------------------------------------------------------------------------
// EdgeConv block: one wave32 per point. X = [center | knn-center] (16 x 2*IN_C)
// in LDS as f16; conv1/conv2 via conv_wmma; conv3 fused with mean-over-K
// (shfl_xor(16) joins the two M-halves) + shortcut + relu, direct to global.
// ---------------------------------------------------------------------------
template <int IN_C, int C>
__global__ void __launch_bounds__(128) edgeconv_kernel(
    const float* __restrict__ fin, const int* __restrict__ idx,
    const _Float16* __restrict__ W1, const float* __restrict__ b1,
    const _Float16* __restrict__ W2, const float* __restrict__ b2,
    const _Float16* __restrict__ W3, const float* __restrict__ b3,
    const _Float16* __restrict__ Wsc, const float* __restrict__ bsc,
    float* __restrict__ fout) {
  constexpr int K2 = 2 * IN_C;
  constexpr size_t XB = (size_t)16 * K2 * sizeof(_Float16);
  constexpr size_t HB = (size_t)16 * C * sizeof(_Float16);
  constexpr bool REUSE = (K2 >= C);  // X buffer reusable as H1
  constexpr size_t PER = ((XB + HB + (REUSE ? 0 : HB)) + 63) & ~(size_t)63;
  extern __shared__ char smem[];
  const int wave = threadIdx.x >> 5;
  const int lane = threadIdx.x & 31;
  char* base = smem + (size_t)wave * PER;
  _Float16* Xs = (_Float16*)base;
  _Float16* H0 = (_Float16*)(base + XB);
  _Float16* H1 = REUSE ? Xs : (_Float16*)(base + XB + HB);

  const int pid = blockIdx.x * 4 + wave;  // grid sized exactly: B*P/4
  const int b = pid / PP;
  const float* fcen = fin + (size_t)pid * IN_C;

  // Build edge-feature matrix X (f16) in LDS.
  for (int m = 0; m < KK; ++m) {
    const int q = idx[(size_t)pid * KK + m];
    const float* fq = fin + ((size_t)b * PP + q) * IN_C;
    for (int c = lane; c < IN_C; c += 32) {
      const float cv = fcen[c];
      Xs[(size_t)m * K2 + c] = (_Float16)cv;
      Xs[(size_t)m * K2 + IN_C + c] = (_Float16)(fq[c] - cv);
    }
  }
  __syncthreads();
  conv_wmma<K2, C>(Xs, W1, b1, H0, lane);
  __syncthreads();
  conv_wmma<C, C>(H0, W2, b2, H1, lane);
  __syncthreads();

  // conv3 + mean over K + shortcut + relu, fused.
  constexpr int KC = C >> 5;
  const int mrow = lane & 15;
  const int hi = lane >> 4;
  for (int nt = 0; nt < C / 16; ++nt) {
    v8f acc = {0.f, 0.f, 0.f, 0.f, 0.f, 0.f, 0.f, 0.f};
    for (int kc = 0; kc < KC; ++kc) {
      const v16h a = load_a_frag<C>(H1, mrow, hi, kc);
      const v16h bm = *(const v16h*)(W3 + ((size_t)(nt * KC + kc) * 32 + lane) * 16);
      acc = __builtin_amdgcn_wmma_f32_16x16x32_f16(false, a, false, bm,
                                                   (short)0, acc, false, false);
    }
    const int n = nt * 16 + mrow;
    const float bn = b3[n];
    float s = 0.f;
#pragma unroll
    for (int i = 0; i < 8; ++i) {
      float v = acc[i] + bn;
      s += v > 0.f ? v : 0.f;
    }
    s += __shfl_xor(s, 16, 32);  // join M 0-7 with M 8-15 (same n)
    if (hi == 0) {
      float sc = bsc[n];
      for (int c = 0; c < IN_C; ++c) sc += fcen[c] * (float)Wsc[(size_t)c * C + n];
      float o = sc + s * (1.f / 16.f);
      fout[(size_t)pid * C + n] = o > 0.f ? o : 0.f;
    }
  }
}

// ---------------------------------------------------------------------------
// Head: masked mean over P, FC(256)+relu, out(6), softmax. One block / batch.
// ---------------------------------------------------------------------------
__global__ void __launch_bounds__(256) head_kernel(
    const float* __restrict__ fts, const float* __restrict__ mask,
    const float* __restrict__ fcW, const float* __restrict__ fcb,
    const float* __restrict__ outW, const float* __restrict__ outb,
    float* __restrict__ out) {
  __shared__ float pool[256];
  __shared__ float h[256];
  __shared__ float lg[6];
  const int b = blockIdx.x, t = threadIdx.x;
  float s = 0.f;
  for (int p = 0; p < PP; ++p)
    s += mask[b * PP + p] * fts[((size_t)b * PP + p) * 256 + t];
  pool[t] = s * (1.f / (float)PP);
  __syncthreads();
  float a = fcb[t];
  for (int c = 0; c < 256; ++c) a += pool[c] * fcW[c * 256 + t];
  h[t] = a > 0.f ? a : 0.f;
  __syncthreads();
  if (t < 6) {
    float l = outb[t];
    for (int c = 0; c < 256; ++c) l += h[c] * outW[c * 6 + t];
    lg[t] = l;
  }
  __syncthreads();
  if (t == 0) {
    float mx = lg[0];
    for (int j = 1; j < 6; ++j) mx = lg[j] > mx ? lg[j] : mx;
    float e[6], sum = 0.f;
    for (int j = 0; j < 6; ++j) { e[j] = expf(lg[j] - mx); sum += e[j]; }
    for (int j = 0; j < 6; ++j) out[b * 6 + j] = e[j] / sum;
  }
}

// ---------------------------------------------------------------------------
static inline size_t ec_shmem(int in_c, int c) {
  const size_t xb = (size_t)16 * 2 * in_c * 2;
  const size_t hb = (size_t)16 * c * 2;
  size_t per = xb + hb + ((2 * in_c >= c) ? 0 : hb);
  per = (per + 63) & ~(size_t)63;
  return per * 4;  // 4 waves per block
}

extern "C" void kernel_launch(void* const* d_in, const int* in_sizes, int n_in,
                              void* d_out, int out_size, void* d_ws, size_t ws_size,
                              hipStream_t stream) {
  (void)in_sizes; (void)n_in; (void)out_size; (void)ws_size;
  const float* FEAT = (const float*)d_in[0];
  auto F32 = [&](int i) { return (const float*)d_in[i]; };

  // Workspace layout (256B aligned chunks).
  char* ws = (char*)d_ws;
  size_t off = 0;
  auto alloc = [&](size_t bytes) -> char* {
    char* r = ws + off;
    off = (off + bytes + 255) & ~(size_t)255;
    return r;
  };
  float* fA  = (float*)alloc((size_t)BB * PP * 256 * 4);
  float* fB  = (float*)alloc((size_t)BB * PP * 256 * 4);
  float* pts = (float*)alloc((size_t)BB * PP * 2 * 4);
  float* msk = (float*)alloc((size_t)BB * PP * 4);
  int*   idx = (int*)  alloc((size_t)BB * PP * KK * 4);
  _Float16* wh = (_Float16*)alloc((size_t)300000 * 2);
  float* bf  = (float*)alloc((size_t)2048 * 4);

  // d_in order (jax pytree, dict keys sorted):
  // 0: features; then edge_convs[0..2]{convs[0..2]{W,bn{beta,gamma,mean,var}},
  // shortcut{W,bn}}, fc{W,b}, input_bn{beta,gamma,mean,var}, out{W,b}.
  const int inc[3] = {16, 64, 128};
  const int cc[3]  = {64, 128, 256};
  size_t Woff[3][4], Boff[3][4];
  size_t wo = 0, bo = 0;
  for (int blk = 0; blk < 3; ++blk) {
    const int Kds[4] = {2 * inc[blk], cc[blk], cc[blk], inc[blk]};
    for (int j = 0; j < 4; ++j) {
      const int Kd = Kds[j], N = cc[blk];
      const int base = 1 + blk * 20 + j * 5;  // W, then bn beta/gamma/mean/var
      Woff[blk][j] = wo;
      Boff[blk][j] = bo;
      prep_weights<<<64, 256, 0, stream>>>(F32(base), F32(base + 1), F32(base + 2),
                                           F32(base + 3), F32(base + 4),
                                           wh + wo, bf + bo, Kd, N, (j < 3) ? 1 : 0);
      wo += (size_t)Kd * N;
      bo += N;
    }
  }

  input_kernel<<<(BB * PP + 255) / 256, 256, 0, stream>>>(
      FEAT, F32(63), F32(64), F32(65), F32(66), pts, msk, fA);

  // Block 0: coords = geometric points (dim 2); fts0 (16ch) in fA -> fB (64ch)
  knn_kernel<2><<<BB, 256, (PP * 2 + 128) * 4, stream>>>(pts, msk, idx);
  edgeconv_kernel<16, 64><<<(BB * PP) / 4, 128, ec_shmem(16, 64), stream>>>(
      fA, idx, wh + Woff[0][0], bf + Boff[0][0], wh + Woff[0][1], bf + Boff[0][1],
      wh + Woff[0][2], bf + Boff[0][2], wh + Woff[0][3], bf + Boff[0][3], fB);

  // Block 1: coords = fts (64ch); fB -> fA (128ch)
  knn_kernel<64><<<BB, 256, (PP * 64 + 128) * 4, stream>>>(fB, msk, idx);
  edgeconv_kernel<64, 128><<<(BB * PP) / 4, 128, ec_shmem(64, 128), stream>>>(
      fB, idx, wh + Woff[1][0], bf + Boff[1][0], wh + Woff[1][1], bf + Boff[1][1],
      wh + Woff[1][2], bf + Boff[1][2], wh + Woff[1][3], bf + Boff[1][3], fA);

  // Block 2: coords = fts (128ch); fA -> fB (256ch)
  knn_kernel<128><<<BB, 256, (PP * 128 + 128) * 4, stream>>>(fA, msk, idx);
  edgeconv_kernel<128, 256><<<(BB * PP) / 4, 128, ec_shmem(128, 256), stream>>>(
      fA, idx, wh + Woff[2][0], bf + Boff[2][0], wh + Woff[2][1], bf + Boff[2][1],
      wh + Woff[2][2], bf + Boff[2][2], wh + Woff[2][3], bf + Boff[2][3], fB);

  head_kernel<<<BB, 256, 0, stream>>>(fB, msk, F32(61), F32(62), F32(67), F32(68),
                                      (float*)d_out);
}
